// MoEGate_2911987826917
// MI455X (gfx1250) — matrix operand order repaired and checked by reading, not compiled, not run on hardware
//
#include <hip/hip_runtime.h>
#include <hip/hip_bf16.h>
#include <math.h>

typedef __attribute__((ext_vector_type(16))) __bf16 v16bf;
typedef __attribute__((ext_vector_type(8)))  float  v8f;
typedef int v4i __attribute__((vector_size(16)));
typedef __attribute__((address_space(1))) v4i* gv4i_p;
typedef __attribute__((address_space(3))) v4i* lv4i_p;

namespace {
constexpr int   kTokens = 8192;
constexpr int   kHid    = 7168;
constexpr int   kExp    = 256;
constexpr int   kGrp    = 8;
constexpr int   kPerG   = 32;
constexpr int   kTopK   = 8;
constexpr int   kTopG   = 4;
constexpr float kScale  = 2.5f;

constexpr int MB   = 64;          // tokens per block
constexpr int NT   = 512;         // threads per block (16 waves)
constexpr int KC   = 128;         // K chunk (f32 elements)
constexpr int APAD = KC + 8;      // 136 f32: LDS row stride for A chunk (bank skew)
constexpr int SPAD = kExp + 8;    // 264 f32: LDS row stride for score matrix
constexpr int ABUF = MB * APAD;   // 8704 f32 per A buffer
constexpr int NCH  = kHid / KC;   // 56 K-chunks
}

#if __has_builtin(__builtin_amdgcn_global_load_async_to_lds_b128)
#define USE_ASYNC 1
#else
#define USE_ASYNC 0
#endif

__device__ __forceinline__ void wait_async_zero() {
#if __has_builtin(__builtin_amdgcn_s_wait_asynccnt)
  __builtin_amdgcn_s_wait_asynccnt(0);
#else
  asm volatile("s_wait_asynccnt 0" ::: "memory");
#endif
}

__device__ __forceinline__ v16bf cvt16(const float4& t0, const float4& t1,
                                       const float4& t2, const float4& t3) {
  v16bf r;
  r[0]=(__bf16)t0.x;  r[1]=(__bf16)t0.y;  r[2]=(__bf16)t0.z;  r[3]=(__bf16)t0.w;
  r[4]=(__bf16)t1.x;  r[5]=(__bf16)t1.y;  r[6]=(__bf16)t1.z;  r[7]=(__bf16)t1.w;
  r[8]=(__bf16)t2.x;  r[9]=(__bf16)t2.y;  r[10]=(__bf16)t2.z; r[11]=(__bf16)t2.w;
  r[12]=(__bf16)t3.x; r[13]=(__bf16)t3.y; r[14]=(__bf16)t3.z; r[15]=(__bf16)t3.w;
  return r;
}

__global__ __launch_bounds__(NT) void moe_gate_kernel(
    const float* __restrict__ x, const float* __restrict__ wgt,
    const float* __restrict__ bias, float* __restrict__ wout,
    int* __restrict__ iout) {
  // smem: phase 1 = double-buffered A chunk (2*8704 f32 = 69.6 KB);
  //       phase 2 = aliased score matrix [MB][SPAD] (16896 f32 <= 17408)
  __shared__ __align__(16) float smem[2 * ABUF];
  __shared__ float gsc[MB][kGrp];

  const int tid  = threadIdx.x;
  const int lane = tid & 31;
  const int wave = tid >> 5;     // 0..15 -> one 16-expert N-tile each
  const int hl   = lane >> 4;    // half-lane
  const int r    = lane & 15;
  const int tokbase = blockIdx.x * MB;

  // W row pointer for this wave's N-tile (expert 16*wave + r)
  const float* wp = wgt + (size_t)(16 * wave + r) * kHid;

  v8f acc[4];   // 4 M-subtiles x 1 N-tile
#pragma unroll
  for (int m = 0; m < 4; ++m)
#pragma unroll
    for (int e = 0; e < 8; ++e) acc[m][e] = 0.0f;

  // Staging map: 512 threads copy MB x KC f32 (32 KB): 16 f32 per thread.
  const int srow = tid >> 3;  // 0..63 token row
  const int sseg = tid & 7;   // 8 segments of 16 f32
  const float* gsrc = x + (size_t)(tokbase + srow) * kHid + sseg * 16;
  const int    ldst = srow * APAD + sseg * 16;

  auto stage_async = [&](int c, int buf) {
#if USE_ASYNC
    const float* g = gsrc + (size_t)c * KC;
    float* l = &smem[buf * ABUF + ldst];
#pragma unroll
    for (int i = 0; i < 4; ++i) {
      __builtin_amdgcn_global_load_async_to_lds_b128(
          (gv4i_p)(g + i * 4), (lv4i_p)(l + i * 4), 0, 0);
    }
#else
    (void)c; (void)buf;
#endif
  };
  auto stage_sync = [&](int c, int buf) {
    const float4* g = (const float4*)(gsrc + (size_t)c * KC);
    float4* l = (float4*)&smem[buf * ABUF + ldst];
#pragma unroll
    for (int i = 0; i < 4; ++i) l[i] = g[i];
  };

  auto compute = [&](int buf, int c) {
    const float* Ab = &smem[buf * ABUF];
#pragma unroll
    for (int kk = 0; kk < KC; kk += 32) {
      // Issue the global B loads first (longest latency), then all LDS A
      // loads, into independent temps so the loads can be clause-batched.
      const float* bp = wp + (size_t)c * KC + kk + 8 * hl;
      const float4 b0 = *(const float4*)(bp);
      const float4 b1 = *(const float4*)(bp + 4);
      const float4 b2 = *(const float4*)(bp + 16);
      const float4 b3 = *(const float4*)(bp + 20);
      float4 ta[4][4];
#pragma unroll
      for (int mt = 0; mt < 4; ++mt) {
        const float* ap = Ab + (16 * mt + r) * APAD + kk + 8 * hl;
        ta[mt][0] = *(const float4*)(ap);
        ta[mt][1] = *(const float4*)(ap + 4);
        ta[mt][2] = *(const float4*)(ap + 16);
        ta[mt][3] = *(const float4*)(ap + 20);
      }
      const v16bf bv = cvt16(b0, b1, b2, b3);
#pragma unroll
      for (int mt = 0; mt < 4; ++mt) {
        const v16bf av = cvt16(ta[mt][0], ta[mt][1], ta[mt][2], ta[mt][3]);
        acc[mt] = __builtin_amdgcn_wmma_f32_16x16x32_bf16(
            false, av, false, bv, (short)0, acc[mt], false, false);
      }
    }
  };

#if USE_ASYNC
  stage_async(0, 0);
  for (int c = 0; c < NCH; ++c) {
    wait_async_zero();          // chunk c resident (this wave's async copies)
    __syncthreads();            // ...and everyone else's
    if (c + 1 < NCH) stage_async(c + 1, (c + 1) & 1);  // overlap with compute
    compute(c & 1, c);
    __syncthreads();            // readers done before buffer is overwritten
  }
#else
  for (int c = 0; c < NCH; ++c) {
    stage_sync(c, c & 1);
    __syncthreads();
    compute(c & 1, c);
    __syncthreads();
  }
#endif

  // ---- Phase 2: sigmoid + bias -> LDS score matrix (aliases A buffers) ----
  // C layout: element v of acc -> M = v + 8*(lane>=16), N = lane&15.
  {
    const int ebase = 16 * wave;
    const float bb = bias[ebase + r];
#pragma unroll
    for (int mt = 0; mt < 4; ++mt) {
#pragma unroll
      for (int v = 0; v < 8; ++v) {
        const int tl = 16 * mt + v + 8 * hl;
        const float s = 1.0f / (1.0f + __expf(-acc[mt][v]));
        smem[tl * SPAD + ebase + r] = s + bb;  // scores_for_choice
      }
    }
  }
  __syncthreads();

  // ---- Phase 3: per (token, group) top-2 sum (512 items, 1/thread) ----
  {
    const int tk = tid >> 3;
    const int g  = tid & 7;
    const float* sp = &smem[tk * SPAD + g * kPerG];
    float m1 = -__builtin_inff(), m2 = -__builtin_inff();
    for (int j = 0; j < kPerG; ++j) {
      const float v = sp[j];
      if (v > m1) { m2 = m1; m1 = v; }
      else if (v > m2) { m2 = v; }
    }
    gsc[tk][g] = m1 + m2;
  }
  __syncthreads();

  // ---- Phase 4: per-token group top-4 + expert top-8, normalize ----
  if (tid < MB) {
    const int tk = tid;
    float gs[kGrp];
#pragma unroll
    for (int g = 0; g < kGrp; ++g) gs[g] = gsc[tk][g];

    int keep = 0;
#pragma unroll
    for (int t = 0; t < kTopG; ++t) {
      float best = -__builtin_inff();
      int bg = 0;
#pragma unroll
      for (int g = 0; g < kGrp; ++g)
        if (!((keep >> g) & 1) && gs[g] > best) { best = gs[g]; bg = g; }
      keep |= 1 << bg;
    }

    float wv[kTopK]; int ei[kTopK];
    float sum = 0.0f;
    float* srow2 = &smem[tk * SPAD];
    for (int k = 0; k < kTopK; ++k) {
      float best = -__builtin_inff();
      int be = 0;
      for (int g = 0; g < kGrp; ++g) {
        if (!((keep >> g) & 1)) continue;
        const int base = g * kPerG;
#pragma unroll 4
        for (int j = 0; j < kPerG; ++j) {
          const float v = srow2[base + j];
          if (v > best) { best = v; be = base + j; }
        }
      }
      srow2[be] = -__builtin_inff();       // destructive mask for next pass
      const float ow = best - bias[be];    // recover original sigmoid score
      wv[k] = ow; ei[k] = be; sum += ow;
    }
    const float nrm = kScale / (sum + 1e-20f);
    const int gt = tokbase + tk;
#pragma unroll
    for (int k = 0; k < kTopK; ++k) {
      wout[(size_t)gt * kTopK + k] = wv[k] * nrm;
      iout[(size_t)gt * kTopK + k] = ei[k];
    }
  }
}

extern "C" void kernel_launch(void* const* d_in, const int* in_sizes, int n_in,
                              void* d_out, int out_size, void* d_ws, size_t ws_size,
                              hipStream_t stream) {
  (void)in_sizes; (void)n_in; (void)out_size; (void)d_ws; (void)ws_size;
  const float* x    = (const float*)d_in[0];
  const float* wgt  = (const float*)d_in[1];
  const float* bias = (const float*)d_in[2];
  float* wout = (float*)d_out;
  int*   iout = (int*)((float*)d_out + (size_t)kTokens * kTopK);

  dim3 grid(kTokens / MB);  // 128 blocks
  dim3 block(NT);           // 16 waves
  hipLaunchKernelGGL(moe_gate_kernel, grid, block, 0, stream,
                     x, wgt, bias, wout, iout);
}
